// ObjectRepresentationSNN_27848567947914
// MI455X (gfx1250) — matrix operand order repaired and checked by reading, not compiled, not run on hardware
//
#include <hip/hip_runtime.h>
#include <hip/hip_bf16.h>
#include <math.h>

// ---------------------------------------------------------------------------
// ObjectRepresentationSNN on MI455X (gfx1250, wave32, WMMA)
//
// Structure exploited (exact in IEEE f32):
//  * steps 1-4: affinity=I, alpha=0 -> inter == 0 -> pure elementwise update
//  * after readouts: M2 == 0 identically, M1 == binary spike-XOR matrix
//    scaled by a per-batch scalar m1 in {0,1}
// Hot loop: per step (5..16), per batch: P = M1 @ [S|C]  (1024x1024 @ 1024x64)
// via v_wmma_f32_16x16x32_f16. A-fragments synthesized from a 32-word spike
// bitmask (no A memory traffic at all); B staged transposed through
// double-buffered LDS, overlapped with the WMMAs.
// ---------------------------------------------------------------------------

typedef __attribute__((ext_vector_type(16))) _Float16 v16h;
typedef __attribute__((ext_vector_type(8)))  float    v8f;

#define NB      8
#define NNODE   1024
#define ND      32
#define DT_C    0.1f
#define ATTR_C  0.5f
#define DECAY_C 0.9f
#define THRESH_C 0.5f
#define INVN    (1.0f/1024.0f)
#define BND     (NB*NNODE*ND)   // 262144
#define LDSPITCH 40             // halves per column; 80B => 16B-aligned b128

// ---------------- workspace layout (bytes) ----------------
#define OFF_THETA  ((size_t)0)
#define OFF_TDEL   ((size_t)1<<20)
#define OFF_GAMMA  ((size_t)2<<20)
#define OFF_SF     ((size_t)3<<20)
#define OFF_CF     ((size_t)4<<20)
#define OFF_SC     ((size_t)5<<20)          // f16 [B][N][64] (S|C), 1MB
#define OFF_MEM    ((size_t)6<<20)
#define OFF_SPK    (((size_t)6<<20)+32768)
#define OFF_ACC    (((size_t)6<<20)+65536)
#define OFF_MASK   (((size_t)6<<20)+98304)
#define OFF_CNT    (((size_t)6<<20)+99328)
#define OFF_M1     (((size_t)6<<20)+99392)

// ---------------------------------------------------------------------------
// Fused init: gamma0 = tanh(|x*W_init + b_init| * gain); then 4 identity-phase
// Kuramoto steps (inter==0): theta += DT*(omega + ATTR*gamma), saving the
// step-2 state as the delayed theta. Also zeroes membrane/spike/trace state.
// ---------------------------------------------------------------------------
__global__ void k_init(const float* __restrict__ x, const float* __restrict__ theta0,
                       const float* __restrict__ W_init, const float* __restrict__ b_init,
                       const float* __restrict__ gain, const float* __restrict__ omega,
                       float* __restrict__ theta, float* __restrict__ tdel,
                       float* __restrict__ gamma, float* __restrict__ membrane,
                       float* __restrict__ spikesF, float* __restrict__ accum) {
  int gid = blockIdx.x * blockDim.x + threadIdx.x;     // [0, BND)
  int b = gid >> 15, rem = gid & 32767;
  int n = rem >> 5,  d = rem & 31;
  float g0 = tanhf(fabsf(x[b*NNODE + n] * W_init[d] + b_init[d]) * gain[n*ND + d]);
  gamma[gid] = g0;
  float u = omega[n*ND + d] + ATTR_C * g0;
  float t = theta0[gid];
  t += DT_C * u;              // step 1
  t += DT_C * u;              // step 2
  tdel[gid] = t;              // delayed theta for step-4 readout
  t += DT_C * u;              // step 3
  t += DT_C * u;              // step 4
  theta[gid] = t;
  if (d == 0) {
    int bn = b*NNODE + n;
    membrane[bn] = 0.0f; spikesF[bn] = 0.0f; accum[bn] = 0.0f;
  }
}

// ---------------------------------------------------------------------------
// Per-step prep: Sf/Cf = sin/cos(theta) in f32 for the epilogue, plus an
// interleaved f16 operand SC[b][n][0..63] = [sin | cos] feeding WMMA-B.
// ---------------------------------------------------------------------------
__global__ void k_prep(const float* __restrict__ theta,
                       float* __restrict__ Sf, float* __restrict__ Cf,
                       _Float16* __restrict__ SC) {
  int gid = blockIdx.x * blockDim.x + threadIdx.x;
  int bn = gid >> 5, d = gid & 31;
  float t = theta[gid];
  float s = sinf(t), c = cosf(t);
  Sf[gid] = s; Cf[gid] = c;
  _Float16* row = SC + (size_t)bn * 64;
  row[d]      = (_Float16)s;
  row[d + 32] = (_Float16)c;
}

// ---------------------------------------------------------------------------
// Stage one 32(K) x 64(col) f16 tile, transposed, into LDS [col][k].
// src is contiguous [k][col]; 128 threads, 2x uint4 loads + 16 b16 stores each.
// ---------------------------------------------------------------------------
__device__ __forceinline__ void stageTile(_Float16* __restrict__ dst,
                                          const _Float16* __restrict__ src,
                                          int tid) {
  #pragma unroll
  for (int i = 0; i < 2; ++i) {
    int u  = tid + i * 128;              // 256 uint4 units of the 2048-half tile
    int k  = u >> 3;                     // row (8 uint4 per 64-half row)
    int c0 = (u & 7) * 8;                // starting column
    union { uint4 q; _Float16 h[8]; } w;
    w.q = *reinterpret_cast<const uint4*>(src + k*64 + c0);
    #pragma unroll
    for (int j = 0; j < 8; ++j) dst[(c0 + j)*LDSPITCH + k] = w.h[j];
  }
}

// ---------------------------------------------------------------------------
// WMMA Kuramoto step: inter = C*(M1@S) - S*(M1@C), M2 == 0.
// Block: 128 threads (4 waves). Wave w owns 16 rows; block owns 64 rows.
// acc[0..1] = (M1@S) cols 0-31, acc[2..3] = (M1@C) cols 0-31.
// ---------------------------------------------------------------------------
__global__ void k_gemm(const _Float16* __restrict__ SC,
                       const float* __restrict__ Sf, const float* __restrict__ Cf,
                       float* __restrict__ theta, float* __restrict__ tdel,
                       const float* __restrict__ gamma, const float* __restrict__ omega,
                       const unsigned* __restrict__ maskp, const float* __restrict__ m1f,
                       int saveDelay) {
  __shared__ _Float16 sB[2][64 * LDSPITCH];   // double-buffered B tiles

  const int b    = blockIdx.y;
  const int lane = threadIdx.x & 31;
  const int wave = threadIdx.x >> 5;
  const int m0   = blockIdx.x * 64 + wave * 16;

  const _Float16* SCb = SC + (size_t)b * NNODE * 64;

  // spike bits: per-lane A-row bit (fixed) + all 32 K-words (one per lane)
  const int row = m0 + (lane & 15);
  const unsigned rw = maskp[b*32 + (row >> 5)];
  const unsigned xorAll = ((rw >> (row & 31)) & 1u) ? 0xFFFFFFFFu : 0u;
  const unsigned kmAll  = maskp[b*32 + lane];       // word `lane` of the mask
  const _Float16 m1h = (_Float16)m1f[b];
  const _Float16 zh  = (_Float16)0.0f;

  v8f acc[4] = {};   // [S cols 0-15][S 16-31][C 0-15][C 16-31]

  union AFrag { v16h v; _Float16 h[16]; };
  union BFrag { v16h v; uint4 q[2]; };

  const int kbase = (lane < 16) ? 0 : 8;
  const int koff  = (lane < 16) ? 0 : 16;
  const int colIn = lane & 15;

  stageTile(sB[0], SCb, threadIdx.x);               // prologue: tile 0

  for (int kb = 0; kb < NNODE/32; ++kb) {
    __syncthreads();                                // tile kb staged
    if (kb + 1 < NNODE/32)                          // overlap: stage kb+1
      stageTile(sB[(kb + 1) & 1], SCb + (size_t)(kb + 1)*32*64, threadIdx.x);

    // ---- synthesize A fragment (16x32 f16) from spike XOR bitmask ----
    unsigned kw = (unsigned)__shfl((int)kmAll, kb); // broadcast word kb
    unsigned xo = kw ^ xorAll;                      // bit j = s_row XOR s_(k0+j)
    AFrag au;
    #pragma unroll
    for (int t = 0; t < 16; ++t) {
      int k = koff + t;                             // contiguous K mapping (k = 16h + t)
      au.h[t] = ((xo >> k) & 1u) ? m1h : zh;
    }

    // ---- load 4 B fragments (32x16 f16) from LDS ----
    const _Float16* buf = sB[kb & 1];
    BFrag bf[4];
    #pragma unroll
    for (int g = 0; g < 4; ++g) {
      int col = g*16 + colIn;
      const uint4* p = reinterpret_cast<const uint4*>(buf + col*LDSPITCH + koff);
      bf[g].q[0] = p[0];
      bf[g].q[1] = p[1];
    }

    // ---- 4x v_wmma_f32_16x16x32_f16 ----
    #pragma unroll
    for (int g = 0; g < 4; ++g) {
      acc[g] = __builtin_amdgcn_wmma_f32_16x16x32_f16(
          false, au.v, false, bf[g].v, (short)0, acc[g], false, false);
    }
  }

  // ---- fused epilogue: theta += DT*(omega + inter/N + ATTR*gamma) ----
  const int rowAdd = (lane < 16) ? 0 : 8;
  #pragma unroll
  for (int r = 0; r < 8; ++r) {
    int i = m0 + r + rowAdd;
    #pragma unroll
    for (int g = 0; g < 2; ++g) {
      int d = colIn + 16*g;
      float ps = acc[g][r];       // (M1 @ S)[i][d]
      float pc = acc[2+g][r];     // (M1 @ C)[i][d]
      size_t off = ((size_t)b*NNODE + i)*ND + d;
      float s = Sf[off], c = Cf[off];
      float inter = c*ps - s*pc;  // M2 contribution is exactly zero
      float th = theta[off] + DT_C*(omega[i*ND + d] + INVN*inter + ATTR_C*gamma[off]);
      theta[off] = th;
      if (saveDelay) tdel[off] = th;
    }
  }
}

// ---------------------------------------------------------------------------
// Readout part 1: gamma = tanh(|theta @ W_read^T + b_read| * gain)
// Also zeroes the per-batch spike bitmask / counters for part 2.
// ---------------------------------------------------------------------------
__global__ void k_gamma(const float* __restrict__ theta, const float* __restrict__ W_read,
                        const float* __restrict__ b_read, const float* __restrict__ gain,
                        float* __restrict__ gamma, unsigned* __restrict__ maskp,
                        int* __restrict__ cnt) {
  int gid = blockIdx.x * blockDim.x + threadIdx.x;
  int b = gid >> 15, rem = gid & 32767;
  int n = rem >> 5,  d = rem & 31;
  const float* th = theta + ((size_t)b*NNODE + n)*ND;
  float a = b_read[d];
  #pragma unroll
  for (int k = 0; k < ND; ++k) a += th[k] * W_read[d*ND + k];
  gamma[gid] = tanhf(fabsf(a) * gain[n*ND + d]);
  if (d == 0) {
    if (n < 32) maskp[b*32 + n] = 0u;
    if (n == 0) cnt[b] = 0;
  }
}

// ---------------------------------------------------------------------------
// Readout part 2: gate/current/leaky membrane/spikes; bitmask + trace accum.
// ---------------------------------------------------------------------------
__global__ void k_spike(const float* __restrict__ gamma, const float* __restrict__ tdel,
                        const float* __restrict__ w_in, const float* __restrict__ b_in,
                        float* __restrict__ membrane, float* __restrict__ spikesF,
                        float* __restrict__ accum, unsigned* __restrict__ maskp,
                        int* __restrict__ cnt, float weight) {
  int g = blockIdx.x * blockDim.x + threadIdx.x;   // [0, NB*NNODE)
  int b = g >> 10, n = g & 1023;
  size_t off3 = ((size_t)b*NNODE + n)*ND;
  float cur = b_in[0];
  #pragma unroll
  for (int d = 0; d < ND; ++d) {
    float gate = 0.5f * (1.0f + sinf(tdel[off3 + d]));
    cur += gate * gamma[off3 + d] * w_in[d];
  }
  float mem = membrane[g];
  float so  = spikesF[g];
  mem = DECAY_C * mem * (1.0f - so) + cur;
  float sn = (mem >= THRESH_C) ? 1.0f : 0.0f;
  membrane[g] = mem;
  spikesF[g]  = sn;
  accum[g]   += weight * sn;
  atomicOr(&maskp[b*32 + (n >> 5)], ((unsigned)sn) << (n & 31));
  if (sn != 0.0f) atomicAdd(&cnt[b], 1);
}

// ---------------------------------------------------------------------------
// Per-batch affinity scalar. Binary delta => aff in {0,1}:
//   mixed batch: aff1 = log1p(1)/max(log1p(1),1e-6) = 1 exactly
//   m1 = aff1*cos(0.5*(1-aff1));  m2 = aff1*sin(0.5*(1-aff1)) == 0 always
// ---------------------------------------------------------------------------
__global__ void k_m1(const int* __restrict__ cnt, float* __restrict__ m1f) {
  int t = threadIdx.x;
  if (t < NB) {
    int c = cnt[t];
    float aff1 = (c > 0 && c < NNODE)
               ? (log1pf(1.0f) / fmaxf(log1pf(1.0f), 1e-6f)) : 0.0f;
    m1f[t] = aff1 * cosf(0.5f * (1.0f - aff1));
  }
}

// ---------------------------------------------------------------------------
// logits = (accum/16) @ W_cls + b_cls   (tiny: 8x10 outputs)
// ---------------------------------------------------------------------------
__global__ void k_logits(const float* __restrict__ accum, const float* __restrict__ W_cls,
                         const float* __restrict__ b_cls, float* __restrict__ out) {
  int t = threadIdx.x;
  if (t < NB*10) {
    int b = t / 10, c = t % 10;
    float a = 0.0f;
    for (int n = 0; n < NNODE; ++n) a += accum[b*NNODE + n] * W_cls[n*10 + c];
    out[t] = a * (1.0f/16.0f) + b_cls[c];
  }
}

// ---------------------------------------------------------------------------
extern "C" void kernel_launch(void* const* d_in, const int* in_sizes, int n_in,
                              void* d_out, int out_size, void* d_ws, size_t ws_size,
                              hipStream_t stream) {
  (void)in_sizes; (void)n_in; (void)out_size; (void)ws_size;
  const float* x      = (const float*)d_in[0];
  const float* theta0 = (const float*)d_in[1];
  const float* W_init = (const float*)d_in[2];
  const float* b_init = (const float*)d_in[3];
  const float* W_read = (const float*)d_in[4];
  const float* b_read = (const float*)d_in[5];
  const float* gain   = (const float*)d_in[6];
  const float* omega  = (const float*)d_in[7];
  const float* w_in   = (const float*)d_in[8];
  const float* b_in   = (const float*)d_in[9];
  const float* W_cls  = (const float*)d_in[10];
  const float* b_cls  = (const float*)d_in[11];
  float* out = (float*)d_out;

  char* ws = (char*)d_ws;
  float*     theta = (float*)(ws + OFF_THETA);
  float*     tdel  = (float*)(ws + OFF_TDEL);
  float*     gamma = (float*)(ws + OFF_GAMMA);
  float*     Sf    = (float*)(ws + OFF_SF);
  float*     Cf    = (float*)(ws + OFF_CF);
  _Float16*  SC    = (_Float16*)(ws + OFF_SC);
  float*     memb  = (float*)(ws + OFF_MEM);
  float*     spk   = (float*)(ws + OFF_SPK);
  float*     acc   = (float*)(ws + OFF_ACC);
  unsigned*  maskp = (unsigned*)(ws + OFF_MASK);
  int*       cnt   = (int*)(ws + OFF_CNT);
  float*     m1f   = (float*)(ws + OFF_M1);

  const int EB = 256, EG = BND / EB;       // elementwise: 1024 blocks x 256

  // steps 1..4 (identity affinity => inter==0), saves delayed theta (step 2)
  k_init<<<EG, EB, 0, stream>>>(x, theta0, W_init, b_init, gain, omega,
                                theta, tdel, gamma, memb, spk, acc);
  // readout @ step 4 (spike weight 4: trace entries for steps 4,5,6,7)
  k_gamma<<<EG, EB, 0, stream>>>(theta, W_read, b_read, gain, gamma, maskp, cnt);
  k_spike<<<NB*NNODE/EB, EB, 0, stream>>>(gamma, tdel, w_in, b_in, memb, spk,
                                          acc, maskp, cnt, 4.0f);
  k_m1<<<1, 32, 0, stream>>>(cnt, m1f);

  dim3 ggrid(NNODE/64, NB);                // 16 x 8 blocks, 128 thr (4 waves)
  for (int seg = 0; seg < 3; ++seg) {      // steps 5-8, 9-12, 13-16
    for (int sub = 0; sub < 4; ++sub) {
      k_prep<<<EG, EB, 0, stream>>>(theta, Sf, Cf, SC);
      k_gemm<<<ggrid, 128, 0, stream>>>(SC, Sf, Cf, theta, tdel, gamma,
                                        omega, maskp, m1f, (sub == 1) ? 1 : 0);
    }
    float w = (seg == 2) ? 1.0f : 4.0f;    // step 16 appended only once
    k_gamma<<<EG, EB, 0, stream>>>(theta, W_read, b_read, gain, gamma, maskp, cnt);
    k_spike<<<NB*NNODE/EB, EB, 0, stream>>>(gamma, tdel, w_in, b_in, memb, spk,
                                            acc, maskp, cnt, w);
    if (seg < 2) k_m1<<<1, 32, 0, stream>>>(cnt, m1f);
  }

  k_logits<<<1, 128, 0, stream>>>(acc, W_cls, b_cls, out);
}
